// MultiAttention_35665408426004
// MI455X (gfx1250) — matrix operand order repaired and checked by reading, not compiled, not run on hardware
//
#include <hip/hip_runtime.h>
#include <hip/hip_bf16.h>
#include <stdint.h>

// ---------------------------------------------------------------------------
// Multi-head attention forward, MI455X (gfx1250), wave32 + WMMA bf16.
//  - Weights (268MB) written to HBM exactly once, non-temporal, never re-read:
//    fused scores->softmax->context kernel keeps the 16-row score stripe in
//    the 320KB WGP LDS.
//  - Every WMMA operand is a contiguous 32B load: projection weights are
//    pre-transposed to bf16 [N][K], and the V projection stores transposed
//    (Vt[bh][d][seq]) so P x V fragments are contiguous too.
//  - Softmax passes are 128-bit vectorized (ds_load_b128 / NT b128 stores).
// HBM traffic ~300MB (~13us at 23.3TB/s); ~30 GFLOP bf16 WMMA is negligible.
// ---------------------------------------------------------------------------

typedef __bf16 bf16_t;
typedef __attribute__((ext_vector_type(16))) __bf16 v16bf;
typedef __attribute__((ext_vector_type(8)))  __bf16 v8bf;
typedef __attribute__((ext_vector_type(4)))  __bf16 v4bf;
typedef __attribute__((ext_vector_type(8)))  float  v8f;
typedef __attribute__((ext_vector_type(4)))  float  v4f;   // native vec for NT stores

#define D_MODEL 768
#define N_HEADS 8
#define DK      64
#define HD      (N_HEADS * DK)   // 512
#define BS      2
#define SEQ     2048
#define ROWS    (BS * SEQ)       // 4096

// LDS row strides (padded so consecutive rows land on different banks)
#define SSTR (SEQ + 4)           // fp32 scores stripe stride (2052, 16B-aligned rows)
#define PSTR (SEQ + 8)           // bf16 probs stripe stride  (2056, 16B-aligned rows)
#define FUSED_LDS_BYTES (16 * SSTR * 4 + 16 * PSTR * 2 + 4 * 16 * 16 * 4)

// ---- fragment layouts (ISA 7.12.2, wave32) --------------------------------
// A (16x32 bf16, MxK): lane L holds row M=L&15; kh=L>>4:
//   a[0..7]  = A[M][kbase + kh*16 + i],  a[8..15] = A[M][kbase + kh*16 + 8 + i]
// B (32x16 bf16, KxN): lane L holds col N=L&15; b[i] = B[kbase + kh*16 + i][N] (as A; not ISA table order)
//   -> if B is stored transposed ([N][K] contiguous), this is ONE 32B load.
// C/D (16x16 f32): acc[r] = C[(L>>4)*8 + r][L&15]

static __device__ inline v16bf load_a_f32(const float* __restrict__ row,
                                          int kbase, int kh) {
  const float* p0 = row + kbase + kh * 16;
  const float* p1 = p0 + 8;
  float4 x0 = *(const float4*)(p0);
  float4 x1 = *(const float4*)(p0 + 4);
  float4 y0 = *(const float4*)(p1);
  float4 y1 = *(const float4*)(p1 + 4);
  v16bf a;
  a[0]  = (__bf16)x0.x; a[1]  = (__bf16)x0.y; a[2]  = (__bf16)x0.z; a[3]  = (__bf16)x0.w;
  a[4]  = (__bf16)x1.x; a[5]  = (__bf16)x1.y; a[6]  = (__bf16)x1.z; a[7]  = (__bf16)x1.w;
  a[8]  = (__bf16)y0.x; a[9]  = (__bf16)y0.y; a[10] = (__bf16)y0.z; a[11] = (__bf16)y0.w;
  a[12] = (__bf16)y1.x; a[13] = (__bf16)y1.y; a[14] = (__bf16)y1.z; a[15] = (__bf16)y1.w;
  return a;
}

static __device__ inline v16bf load_a_bf16(const bf16_t* __restrict__ row,
                                           int kbase, int kh) {
  v8bf lo = *(const v8bf*)(row + kbase + kh * 16);
  v8bf hi = *(const v8bf*)(row + kbase + kh * 16 + 8);
  v16bf a;
#pragma unroll
  for (int i = 0; i < 8; ++i) { a[i] = lo[i]; a[8 + i] = hi[i]; }
  return a;
}

#define WMMA_BF16(a, b, c) \
  __builtin_amdgcn_wmma_f32_16x16x32_bf16(false, (a), false, (b), (short)0, (c), false, false)

// ---------------------------------------------------------------------------
// 0) Transpose+convert weight matrix W[K][N] (f32) -> WT[N][K] (bf16), so all
//    GEMM B-fragments become contiguous 32B loads.
// ---------------------------------------------------------------------------
__global__ void transpose_to_bf16_kernel(const float* __restrict__ W,
                                         bf16_t* __restrict__ WT,
                                         int K, int N) {
  int idx = blockIdx.x * blockDim.x + threadIdx.x;
  if (idx >= K * N) return;
  int n = idx / K, kk = idx - n * K;
  WT[idx] = (__bf16)W[(size_t)kk * N + n];
}

// ---------------------------------------------------------------------------
// 1) Projection: Y[4096x512] (bf16) = X[4096x768] (f32) * W, with W given as
//    WT[N=512][K=768] bf16.  One 16x16 C tile per wave.
// ---------------------------------------------------------------------------
__global__ void proj_kernel(const float* __restrict__ X,
                            const bf16_t* __restrict__ WT,
                            bf16_t* __restrict__ Y) {
  const int tilesN = HD / 16;                      // 32
  int wave = blockIdx.x * (blockDim.x >> 5) + (threadIdx.x >> 5);
  int tM = wave / tilesN, tN = wave % tilesN;
  if (tM >= ROWS / 16) return;                     // wave-uniform
  int lane = threadIdx.x & 31;
  int r = lane & 15, kh = lane >> 4;

  const float*  arow = X + (size_t)(tM * 16 + r) * D_MODEL;
  const bf16_t* brow = WT + (size_t)(tN * 16 + r) * D_MODEL;
  v8f acc = {};
  for (int k0 = 0; k0 < D_MODEL; k0 += 32) {
    v16bf a = load_a_f32(arow, k0, kh);
    v16bf b = *(const v16bf*)(brow + k0 + kh * 16);
    acc = WMMA_BF16(a, b, acc);
  }
  bf16_t* out = Y + (size_t)(tM * 16 + kh * 8) * HD + tN * 16 + r;
#pragma unroll
  for (int i = 0; i < 8; ++i) out[(size_t)i * HD] = (__bf16)acc[i];
}

// ---------------------------------------------------------------------------
// 1b) V projection storing TRANSPOSED result: Vt[bh][d][seq] (bf16).
//     acc[i] runs along seq -> one packed 16B store per lane.
// ---------------------------------------------------------------------------
__global__ void proj_v_kernel(const float* __restrict__ X,
                              const bf16_t* __restrict__ WT,
                              bf16_t* __restrict__ Vt) {
  const int tilesN = HD / 16;                      // 32
  int wave = blockIdx.x * (blockDim.x >> 5) + (threadIdx.x >> 5);
  int tM = wave / tilesN, tN = wave % tilesN;
  if (tM >= ROWS / 16) return;
  int lane = threadIdx.x & 31;
  int r = lane & 15, kh = lane >> 4;

  const float*  arow = X + (size_t)(tM * 16 + r) * D_MODEL;
  const bf16_t* brow = WT + (size_t)(tN * 16 + r) * D_MODEL;
  v8f acc = {};
  for (int k0 = 0; k0 < D_MODEL; k0 += 32) {
    v16bf a = load_a_f32(arow, k0, kh);
    v16bf b = *(const v16bf*)(brow + k0 + kh * 16);
    acc = WMMA_BF16(a, b, acc);
  }
  // C element acc[i]: seq-row m = tM*16 + kh*8 + i, proj-col n = tN*16 + r
  int b   = (tM * 16) / SEQ;
  int s   = (tM * 16) % SEQ + kh * 8;              // 16B-aligned in bf16
  int n   = tN * 16 + r;
  int h   = n >> 6, d = n & 63;
  v8bf pk;
#pragma unroll
  for (int i = 0; i < 8; ++i) pk[i] = (__bf16)acc[i];
  *(v8bf*)(Vt + ((size_t)(b * N_HEADS + h) * DK + d) * SEQ + s) = pk;
}

// ---------------------------------------------------------------------------
// 2) Fused attention core, one workgroup (8 waves) per (bh, 16-row q-tile):
//    A: WMMA scores stripe (16 x 2048) -> LDS fp32 (Q frags loaded once)
//    B: vectorized softmax in LDS; weights -> d_out via NT b128 stores;
//       packed bf16 probs -> LDS
//    C: WMMA context = P(LDS) x Vt(global, contiguous), K split across wave
//       pairs, LDS reduction; prefetch on the streaming Vt operand.
// ---------------------------------------------------------------------------
__global__ void fused_attn_kernel(const bf16_t* __restrict__ Q,
                                  const bf16_t* __restrict__ K,
                                  const bf16_t* __restrict__ Vt,
                                  const unsigned char* __restrict__ mask,
                                  float* __restrict__ Wout,
                                  bf16_t* __restrict__ At) {
  extern __shared__ char smem[];
  float*  sS = (float*)smem;                               // [16][SSTR] fp32
  bf16_t* sP = (bf16_t*)(smem + 16 * SSTR * 4);            // [16][PSTR] bf16
  float*  sC = (float*)(smem + 16 * SSTR * 4 + 16 * PSTR * 2); // [4][16][16]

  const int bh = blockIdx.z, b = bh >> 3, h = bh & 7;
  const int tQ = blockIdx.x;                               // 0..127
  const int wv = threadIdx.x >> 5;                         // 0..7
  const int lane = threadIdx.x & 31;
  const int r = lane & 15, kh = lane >> 4;

  // ---- Phase A: score stripe into LDS --------------------------------------
  const bf16_t* qrow = Q + (size_t)(b * SEQ + tQ * 16 + r) * HD + h * DK;
  const v16bf aq0 = load_a_bf16(qrow, 0, kh);              // reused 128x
  const v16bf aq1 = load_a_bf16(qrow, 32, kh);
  const unsigned char* mbase = mask + (size_t)b * SEQ * SEQ + (size_t)(tQ * 16) * SEQ;

  for (int t = wv; t < SEQ / 16; t += 8) {
    const bf16_t* krow = K + (size_t)(b * SEQ + t * 16 + r) * HD + h * DK;
    v16bf b0 = *(const v16bf*)(krow + kh * 16);
    v16bf b1 = *(const v16bf*)(krow + 32 + kh * 16);
    v8f acc = {};
    acc = WMMA_BF16(aq0, b0, acc);
    acc = WMMA_BF16(aq1, b1, acc);
    const int col = t * 16 + r;
#pragma unroll
    for (int i = 0; i < 8; ++i) {
      float sv = acc[i] * 0.125f;                          // 1/sqrt(64)
      if (mbase[(size_t)(kh * 8 + i) * SEQ + col]) sv = -1e9f; // all-false here
      sS[(kh * 8 + i) * SSTR + col] = sv;
    }
  }
  __syncthreads();

  // ---- Phase B: vectorized softmax; NT-store weights; bf16 probs to LDS ----
  for (int row = wv; row < 16; row += 8) {
    float* p = sS + row * SSTR;
    float m = -3.0e38f;
    for (int i = lane * 4; i < SEQ; i += 128) {
      v4f x = *(const v4f*)(p + i);
      m = fmaxf(m, fmaxf(fmaxf(x.x, x.y), fmaxf(x.z, x.w)));
    }
#pragma unroll
    for (int off = 16; off > 0; off >>= 1) m = fmaxf(m, __shfl_xor(m, off, 32));
    float sum = 0.0f;
    for (int i = lane * 4; i < SEQ; i += 128) {
      v4f x = *(const v4f*)(p + i);
      x.x = __expf(x.x - m); x.y = __expf(x.y - m);
      x.z = __expf(x.z - m); x.w = __expf(x.w - m);
      sum += (x.x + x.y) + (x.z + x.w);
      *(v4f*)(p + i) = x;
    }
#pragma unroll
    for (int off = 16; off > 0; off >>= 1) sum += __shfl_xor(sum, off, 32);
    const float inv = 1.0f / sum;
    float*  wrow = Wout + ((size_t)bh * SEQ + tQ * 16 + row) * SEQ;
    bf16_t* prow = sP + row * PSTR;
    for (int i = lane * 4; i < SEQ; i += 128) {
      v4f x = *(const v4f*)(p + i);
      x *= inv;
      __builtin_nontemporal_store(x, (v4f*)(wrow + i));    // write-once output
      v4bf pk;
      pk[0] = (__bf16)x.x; pk[1] = (__bf16)x.y;
      pk[2] = (__bf16)x.z; pk[3] = (__bf16)x.w;
      *(v4bf*)(prow + i) = pk;
    }
  }
  __syncthreads();

  // ---- Phase C: context tile = P x V, K halves split across wave pairs -----
  const int tN = wv & 3;                                   // d-tile 0..3
  const int krh = wv >> 2;                                 // K half 0/1
  const bf16_t* vrow = Vt + ((size_t)bh * DK + tN * 16 + r) * SEQ;
  v8f acc = {};
  for (int k0 = krh * (SEQ / 2); k0 < (krh + 1) * (SEQ / 2); k0 += 32) {
    const bf16_t* prow = sP + r * PSTR + k0;
    v8bf lo0 = *(const v8bf*)(prow + kh * 16);
    v8bf hi0 = *(const v8bf*)(prow + kh * 16 + 8);
    v16bf a;
#pragma unroll
    for (int i = 0; i < 8; ++i) { a[i] = lo0[i]; a[8 + i] = hi0[i]; }
    v16bf bb = *(const v16bf*)(vrow + k0 + kh * 16);       // contiguous 32B
    __builtin_prefetch(vrow + k0 + 256, 0, 1);             // global_prefetch_b8
    acc = WMMA_BF16(a, bb, acc);
  }
  if (krh == 1) {                                          // wave-uniform branch
#pragma unroll
    for (int i = 0; i < 8; ++i) sC[tN * 256 + (kh * 8 + i) * 16 + r] = acc[i];
  }
  __syncthreads();
  if (krh == 0) {
#pragma unroll
    for (int i = 0; i < 8; ++i) acc[i] += sC[tN * 256 + (kh * 8 + i) * 16 + r];
    bf16_t* outp = At + (size_t)(b * SEQ + tQ * 16 + kh * 8) * HD + h * DK + tN * 16 + r;
#pragma unroll
    for (int i = 0; i < 8; ++i) outp[(size_t)i * HD] = (__bf16)acc[i];
  }
}

// ---------------------------------------------------------------------------
// 3) Output projection: O[4096x768] (f32) = At[4096x512] (bf16) * Wo, with
//    Wo given as WoT[N=768][K=512] bf16.
// ---------------------------------------------------------------------------
__global__ void out_kernel(const bf16_t* __restrict__ At,
                           const bf16_t* __restrict__ WoT,
                           float* __restrict__ O) {
  const int tilesN = D_MODEL / 16;                 // 48
  int wave = blockIdx.x * (blockDim.x >> 5) + (threadIdx.x >> 5);
  int tM = wave / tilesN, tN = wave % tilesN;
  if (tM >= ROWS / 16) return;
  int lane = threadIdx.x & 31;
  int r = lane & 15, kh = lane >> 4;

  const bf16_t* arow = At + (size_t)(tM * 16 + r) * HD;
  const bf16_t* brow = WoT + (size_t)(tN * 16 + r) * HD;
  v8f acc = {};
  for (int k0 = 0; k0 < HD; k0 += 32) {
    v16bf a = load_a_bf16(arow, k0, kh);
    v16bf b = *(const v16bf*)(brow + k0 + kh * 16);
    acc = WMMA_BF16(a, b, acc);
  }
  float* out = O + (size_t)(tM * 16 + kh * 8) * D_MODEL + tN * 16 + r;
#pragma unroll
  for (int i = 0; i < 8; ++i) out[(size_t)i * D_MODEL] = acc[i];
}

// ---------------------------------------------------------------------------
extern "C" void kernel_launch(void* const* d_in, const int* in_sizes, int n_in,
                              void* d_out, int out_size, void* d_ws, size_t ws_size,
                              hipStream_t stream) {
  const float* q    = (const float*)d_in[0];
  const float* k    = (const float*)d_in[1];
  const float* v    = (const float*)d_in[2];
  const unsigned char* mask = (const unsigned char*)d_in[3]; // all-false here
  const float* Wq   = (const float*)d_in[4];
  const float* Wk   = (const float*)d_in[5];
  const float* Wv   = (const float*)d_in[6];
  const float* Wo   = (const float*)d_in[7];

  float* out     = (float*)d_out;                    // (2,2048,768)
  float* weights = out + (size_t)ROWS * D_MODEL;     // (2,8,2048,2048)

  // Workspace (bf16): Yq, Yk (4096x512), Vt (16x64x2048), At (4096x512),
  // WqT, WkT, WvT (512x768), WoT (768x512)  -> ~20 MB total
  bf16_t* Yq  = (bf16_t*)d_ws;
  bf16_t* Yk  = Yq  + (size_t)ROWS * HD;
  bf16_t* Vt  = Yk  + (size_t)ROWS * HD;
  bf16_t* At  = Vt  + (size_t)ROWS * HD;
  bf16_t* WqT = At  + (size_t)ROWS * HD;
  bf16_t* WkT = WqT + (size_t)D_MODEL * HD;
  bf16_t* WvT = WkT + (size_t)D_MODEL * HD;
  bf16_t* WoT = WvT + (size_t)D_MODEL * HD;

  // 0) Weight transpose+convert (tiny: ~16MB traffic once per call)
  {
    int n = D_MODEL * HD;                          // 393216
    dim3 grd((n + 255) / 256), blk(256);
    transpose_to_bf16_kernel<<<grd, blk, 0, stream>>>(Wq, WqT, D_MODEL, HD);
    transpose_to_bf16_kernel<<<grd, blk, 0, stream>>>(Wk, WkT, D_MODEL, HD);
    transpose_to_bf16_kernel<<<grd, blk, 0, stream>>>(Wv, WvT, D_MODEL, HD);
    transpose_to_bf16_kernel<<<grd, blk, 0, stream>>>(Wo, WoT, HD, D_MODEL);
  }
  // 1) Q/K/V projections: 8192 tiles each, 4 waves/block
  {
    dim3 blk(128);
    dim3 grd(((ROWS / 16) * (HD / 16)) / 4);
    proj_kernel  <<<grd, blk, 0, stream>>>(q, WqT, Yq);
    proj_kernel  <<<grd, blk, 0, stream>>>(k, WkT, Yk);
    proj_v_kernel<<<grd, blk, 0, stream>>>(v, WvT, Vt);
  }
  // 2) Fused scores + softmax + context (weights written once, NT)
  {
    dim3 grd(SEQ / 16, 1, BS * N_HEADS);           // 128 x 16
    fused_attn_kernel<<<grd, dim3(256), FUSED_LDS_BYTES, stream>>>(
        Yq, Yk, Vt, mask, weights, At);
  }
  // 3) Output projection
  {
    dim3 grd(((ROWS / 16) * (D_MODEL / 16)) / 4);
    out_kernel<<<grd, dim3(128), 0, stream>>>(At, WoT, out);
  }
}
